// RNN_layers_45595372814875
// MI455X (gfx1250) — matrix operand order repaired and checked by reading, not compiled, not run on hardware
//
#include <hip/hip_runtime.h>
#include <hip/hip_bf16.h>
#include <math.h>

typedef __attribute__((ext_vector_type(16))) _Float16 v16h;
typedef __attribute__((ext_vector_type(8)))  _Float16 v8h;
typedef __attribute__((ext_vector_type(8)))  float    v8f;

// Problem constants (from reference)
constexpr int kB = 8192;
constexpr int kT = 187;
constexpr int kH = 128;
constexpr int kC = 5;

constexpr int kHS = 136;   // padded LDS row stride in halves (272 B -> staggered banks)

// Branchless tanh: prefer the gfx1250 V_TANH_F32 trans op.
#if __has_builtin(__builtin_amdgcn_tanhf)
static __device__ __forceinline__ float fast_tanh(float z) {
    return __builtin_amdgcn_tanhf(z);
}
#else
static __device__ __forceinline__ float fast_tanh(float z) {
    // tanh(z) = 1 - 2/(exp(2z)+1);  exp(2z) = exp2(2*log2e*z)
    const float e = __builtin_amdgcn_exp2f(z * 2.8853900817779268f);
    return 1.0f - 2.0f * __builtin_amdgcn_rcpf(e + 1.0f);
}
#endif

__global__ __launch_bounds__(64)
void rnn_fused_wmma_kernel(const float* __restrict__ x,
                           const float* __restrict__ W,
                           const float* __restrict__ U,
                           const float* __restrict__ b,
                           const float* __restrict__ Wd,
                           const float* __restrict__ bd,
                           float* __restrict__ out)
{
    // U transposed, f16: sUT[n][k] = U[k][n] (staging for register-resident B)
    __shared__ __align__(16) _Float16 sUT[kH * kHS];
    // Double-buffered hidden state h[16 rows][128], f16, shared by the 2 waves
    __shared__ __align__(16) _Float16 sH[2][16 * kHS];

    const int tid  = threadIdx.x;
    const int wave = tid >> 5;    // 0 or 1: owns column tiles 4*wave .. 4*wave+3
    const int lane = tid & 31;
    const int nl   = lane & 15;   // row (A) / col (B, C/D) index within 16
    const int hs   = lane >> 4;   // half-wave select

    // ---- one-time: U (f32 [k][n]) -> sUT (f16 [n][k]) ----
    for (int i = tid; i < kH * kH; i += blockDim.x) {
        const int k = i >> 7;
        const int n = i & (kH - 1);
        sUT[n * kHS + k] = (_Float16)U[k * kH + n];
    }
    // h0 = 0 in read buffer for t=0 (buffer 0)
    for (int i = tid; i < 16 * kHS; i += blockDim.x) sH[0][i] = (_Float16)0.f;
    __syncthreads();

    // ---- one-time: load this wave's 16 B fragments into registers (128 VGPRs) ----
    // B layout (f16 32x16): lane holds column n, K = k0 + 16*hs .. +15 contiguous in sUT row
    v16h bfr[4][4];
#pragma unroll
    for (int cx = 0; cx < 4; ++cx) {
        const int ct = wave * 4 + cx;
#pragma unroll
        for (int kc = 0; kc < 4; ++kc) {
            const int k0 = kc * 32;
            const _Float16* bb = &sUT[(ct * 16 + nl) * kHS + k0 + 16 * hs];
            ((v8h*)&bfr[cx][kc])[0] = *(const v8h*)(bb);
            ((v8h*)&bfr[cx][kc])[1] = *(const v8h*)(bb + 8);
        }
    }

    const int row_base = blockIdx.x * 16;

    // Per-lane W/b for this wave's 4 column tiles (col = (4*wave+cx)*16 + nl)
    float wreg[4], breg[4];
#pragma unroll
    for (int cx = 0; cx < 4; ++cx) {
        wreg[cx] = W[(wave * 4 + cx) * 16 + nl];
        breg[cx] = b[(wave * 4 + cx) * 16 + nl];
    }

    // ---- 187-step recurrence: 1 barrier per step, B operand register-resident ----
    for (int t = 0; t < kT; ++t) {
        const int rb = t & 1;        // read buffer
        const int wb = rb ^ 1;       // write buffer

        // x contribution: lane (j,hs) owns C/D row m = j + 8*hs
        float xr[8];
#pragma unroll
        for (int j = 0; j < 8; ++j) {
            const int m = j + 8 * hs;
            xr[j] = x[(size_t)(row_base + m) * kT + t];
        }

        // Load A fragments: h[16 x 128] as 4 chunks of 16x32 f16.
        // A layout: lanes 0-15 hold K = k0..k0+15; lanes 16-31 hold K = k0+16..k0+31.
        v16h a[4];
#pragma unroll
        for (int kc = 0; kc < 4; ++kc) {
            const int k0 = kc * 32;
            const _Float16* ab = &sH[rb][nl * kHS + k0 + 16 * hs];
            ((v8h*)&a[kc])[0] = *(const v8h*)(ab);
            ((v8h*)&a[kc])[1] = *(const v8h*)(ab + 8);
        }

        // This wave's 4 column tiles: K=128 via 4 WMMAs each, fused tanh epilogue
#pragma unroll
        for (int cx = 0; cx < 4; ++cx) {
            v8f c = {};
#pragma unroll
            for (int kc = 0; kc < 4; ++kc) {
                c = __builtin_amdgcn_wmma_f32_16x16x32_f16(
                        false, a[kc], false, bfr[cx][kc],
                        (short)0, c, false, false);
            }
            const int col = (wave * 4 + cx) * 16 + nl;
#pragma unroll
            for (int j = 0; j < 8; ++j) {
                const int m = j + 8 * hs;
                const float pre = fmaf(xr[j], wreg[cx], breg[cx]);
                const float hn  = fast_tanh(c[j] + pre);
                sH[wb][m * kHS + col] = (_Float16)hn;
            }
        }
        __syncthreads();   // single barrier: publishes buf wb before partner reads it
    }

    // ---- dense head + softmax: threads 0..15 each own one row ----
    const _Float16* fH = &sH[kT & 1][0];   // final state lives in buffer written at t=kT-1
    if (tid < 16) {
        const int row = row_base + tid;
        float logits[kC];
#pragma unroll
        for (int c = 0; c < kC; ++c) logits[c] = bd[c];
        for (int h = 0; h < kH; ++h) {
            const float hv = (float)fH[tid * kHS + h];
#pragma unroll
            for (int c = 0; c < kC; ++c) logits[c] += hv * Wd[h * kC + c];
        }
        float mx = logits[0];
#pragma unroll
        for (int c = 1; c < kC; ++c) mx = fmaxf(mx, logits[c]);
        float e[kC], s = 0.f;
#pragma unroll
        for (int c = 0; c < kC; ++c) { e[c] = __expf(logits[c] - mx); s += e[c]; }
        const float inv = 1.f / s;
#pragma unroll
        for (int c = 0; c < kC; ++c) out[(size_t)row * kC + c] = e[c] * inv;
    }
}

extern "C" void kernel_launch(void* const* d_in, const int* in_sizes, int n_in,
                              void* d_out, int out_size, void* d_ws, size_t ws_size,
                              hipStream_t stream) {
    const float* x  = (const float*)d_in[0];   // [B,T,1]
    const float* W  = (const float*)d_in[1];   // [1,H]
    const float* U  = (const float*)d_in[2];   // [H,H]
    const float* b  = (const float*)d_in[3];   // [H]
    const float* Wd = (const float*)d_in[4];   // [H,C]
    const float* bd = (const float*)d_in[5];   // [C]
    float* out = (float*)d_out;                // [B,C]

    dim3 grid(kB / 16);    // 512 workgroups, 16 batch rows each
    dim3 block(64);        // 2 waves (wave32): column tiles split across waves
    rnn_fused_wmma_kernel<<<grid, block, 0, stream>>>(x, W, U, b, Wd, bd, out);
}
